// MetaSR_86517821211809
// MI455X (gfx1250) — matrix operand order repaired and checked by reading, not compiled, not run on hardware
//
#include <hip/hip_runtime.h>

typedef __attribute__((ext_vector_type(16))) _Float16 v16h;
typedef __attribute__((ext_vector_type(8)))  _Float16 v8h;
typedef __attribute__((ext_vector_type(8)))  float    v8f;

#define HW   128
#define CIN  3
#define CF   64
#define NQ   65536      /* B*Q flattened */
#define HID  256
#define NG   576        /* C*9 */
#define OUT3 1728       /* NG*3 */

/* workspace byte offsets (all 256-aligned) */
#define WS_FEAT 0u          /* f32 [4][64][128][128]  = 16777216 B */
#define WS_HDN  16777216u   /* f16 [65536][256]       = 33554432 B */
#define WS_W2H  50331648u   /* f16 [3][576][256]      =   884736 B */
#define WS_QIDX 51216384u   /* i32 [65536]            =   262144 B */

/* ---------------- kernel 1: 3x3 conv, SAME pad, NCHW ---------------- */
__global__ void __launch_bounds__(256)
conv3x3_kernel(const float* __restrict__ inp, const float* __restrict__ cw,
               const float* __restrict__ cb, float* __restrict__ feat) {
  __shared__ float w[CF * CIN * 9];
  __shared__ float bsh[CF];
  int tid = threadIdx.x;
  for (int i = tid; i < CF * CIN * 9; i += 256) w[i] = cw[i];
  for (int i = tid; i < CF; i += 256) bsh[i] = cb[i];
  __syncthreads();
  int gid = blockIdx.x * 256 + tid;          /* b*HW*HW + y*HW + x */
  int x = gid & (HW - 1);
  int y = (gid >> 7) & (HW - 1);
  int b = gid >> 14;
  float in[27];
#pragma unroll
  for (int ci = 0; ci < CIN; ci++)
#pragma unroll
    for (int dy = 0; dy < 3; dy++)
#pragma unroll
      for (int dx = 0; dx < 3; dx++) {
        int yy = y + dy - 1, xx = x + dx - 1;
        bool ok = ((unsigned)yy < HW) && ((unsigned)xx < HW);
        in[(ci * 3 + dy) * 3 + dx] = ok ? inp[((b * CIN + ci) * HW + yy) * HW + xx] : 0.f;
      }
  for (int co = 0; co < CF; co++) {
    float acc = bsh[co];
#pragma unroll
    for (int k = 0; k < 27; k++) acc = fmaf(w[co * 27 + k], in[k], acc);
    feat[((b * CF + co) * HW + y) * HW + x] = acc;
  }
}

/* ------- kernel 2: transpose fc2_w [256][1728] -> W2h[c][g][k] f16 ------- */
__global__ void __launch_bounds__(256)
prep_w2_kernel(const float* __restrict__ w2, _Float16* __restrict__ w2h) {
  int i = blockIdx.x * 256 + threadIdx.x;    /* i = (c*NG+g)*HID + k */
  if (i >= 3 * NG * HID) return;
  int k = i & (HID - 1);
  int g = (i >> 8) % NG;
  int c = i / (NG * HID);
  w2h[i] = (_Float16)w2[k * OUT3 + g * 3 + c];
}

/* ------- kernel 3: coord math + fc1 + relu -> hdn f16, packed iy|ix ------- */
__global__ void __launch_bounds__(256)
mlp1_kernel(const float* __restrict__ coord, const float* __restrict__ cell,
            const float* __restrict__ w1, const float* __restrict__ b1,
            _Float16* __restrict__ hdn, int* __restrict__ qidx) {
  int qq = blockIdx.x;
  int t = threadIdx.x;
  float cl0 = cell[qq * 2 + 0], cl1 = cell[qq * 2 + 1];
  float c0 = coord[qq * 2 + 0] - 0.5f * cl0;
  float c1 = coord[qq * 2 + 1] - 0.5f * cl1;
  const float eps = 1e-6f;
  float cq0 = fminf(fmaxf(c0 + eps, -1.f + eps), 1.f - eps);
  float cq1 = fminf(fmaxf(c1 + eps, -1.f + eps), 1.f - eps);
  int iy = (int)rintf(((cq0 + 1.f) * (float)HW - 1.f) * 0.5f);
  int ix = (int)rintf(((cq1 + 1.f) * (float)HW - 1.f) * 0.5f);
  iy = min(max(iy, 0), HW - 1);
  ix = min(max(ix, 0), HW - 1);
  float qc0 = -1.f + (2.f / HW) * (float)iy;
  float qc1 = -1.f + (2.f / HW) * (float)ix;
  float m0 = (c0 - qc0) * (HW * 0.5f);
  float m1 = (c1 - qc1) * (HW * 0.5f);
  float m2 = cl0 * (HW * 0.5f);
  float v = fmaf(m0, w1[t], fmaf(m1, w1[HID + t], fmaf(m2, w1[2 * HID + t], b1[t])));
  hdn[qq * HID + t] = (_Float16)fmaxf(v, 0.f);
  if (t == 0) qidx[qq] = iy | (ix << 16);
}

/* ---- per-lane async copy of 16B global -> LDS (ASYNCcnt-tracked) ---- */
__device__ __forceinline__ void async_b128(unsigned lds_off, const void* gptr) {
  asm volatile("global_load_async_to_lds_b128 %0, %1, off"
               :: "v"(lds_off), "v"(gptr) : "memory");
}

/* ------- kernel 4: fused fc2 WMMA GEMM + gather + contract -------
   B tiles staged into LDS once per workgroup via async DMA, double-buffered. */
__global__ void __launch_bounds__(256)
meta_sr_fused_kernel(const _Float16* __restrict__ hdn, const _Float16* __restrict__ w2h,
                     const float* __restrict__ b2, const float* __restrict__ feat,
                     const int* __restrict__ qidx, float* __restrict__ out) {
  __shared__ _Float16 bstage[2][3][16][HID];   /* 2 x 24 KB */
  const int tid = threadIdx.x;
  const int lane = tid & 31;
  const int wave = tid >> 5;
  const int q0 = (blockIdx.x * 8 + wave) * 16; /* 16 queries per wave */
  const int ln = lane & 15;
  const int hi = lane >> 4;                    /* 0 or 1 */
  const unsigned lds0 = (unsigned)(uintptr_t)(&bstage[0][0][0][0]);

  /* cooperative async stage of one g-tile's B data: [c][gl][k] f16, 24 KB.
     thread t copies 16B chunks i = t, t+256, ..., t+1280 (6 per thread). */
  auto issue_tile = [&](int gt, int buf) {
    unsigned dst = lds0 + (unsigned)buf * 24576u;
#pragma unroll
    for (int j = 0; j < 6; j++) {
      int i = tid + j * 256;                   /* 0..1535 */
      int c = i >> 9;                          /* 512 chunks of 16B per c */
      int rem = i & 511;
      const _Float16* src = w2h + (size_t)(c * NG + gt * 16) * HID + rem * 8;
      async_b128(dst + (unsigned)i * 16u, (const void*)src);
    }
  };

  /* ---- A tile: hdn[16 queries][K=256] f16, ISA 16-bit A layout ----
     lane<16: M=lane, halves -> K {0..7,16..23}+32*ks; lane>=16: M=lane-16, +8 */
  const _Float16* arow = hdn + (size_t)(q0 + ln) * HID + hi * 8;
  v16h A[8];
#pragma unroll
  for (int ks = 0; ks < 8; ks++) {
    v8h lo = *(const v8h*)(arow + ks * 32);
    v8h hh = *(const v8h*)(arow + ks * 32 + 16);
    A[ks] = __builtin_shufflevector(lo, hh, 0, 1, 2, 3, 4, 5, 6, 7,
                                    8, 9, 10, 11, 12, 13, 14, 15);
  }

  /* per-D-row query state (D rows: M = r + 8*hi) */
  int iyr[8], ixr[8], bO[8];
#pragma unroll
  for (int r = 0; r < 8; r++) {
    int qD = q0 + r + hi * 8;
    int p = qidx[qD];
    iyr[r] = p & 0xffff;
    ixr[r] = p >> 16;
    bO[r] = (qD >> 14) * (CF * HW * HW);       /* batch offset into feat */
  }

  float part[3][8];
#pragma unroll
  for (int c = 0; c < 3; c++)
#pragma unroll
    for (int r = 0; r < 8; r++) part[c][r] = 0.f;

  issue_tile(0, 0);                            /* prologue: tile 0 -> buf 0 */

  for (int gt = 0; gt < NG / 16; gt++) {
    const int cur = gt & 1;
    if (gt < NG / 16 - 1) {
      issue_tile(gt + 1, cur ^ 1);             /* stream next tile */
      asm volatile("s_wait_asynccnt 0x6" ::: "memory");  /* cur tile landed */
    } else {
      asm volatile("s_wait_asynccnt 0x0" ::: "memory");
    }
    __syncthreads();                           /* all waves' chunks visible */

    int gg = gt * 16 + ln;                     /* this lane's N column = g */
    int ch = gg / 9;
    int j = gg - ch * 9;
    int ky = j / 3 - 1;
    int kx = j - (j / 3) * 3 - 1;
    const float* fch = feat + ch * HW * HW;

    /* gather q_feat column for the 8 D rows this lane owns */
    float qf[8];
#pragma unroll
    for (int r = 0; r < 8; r++) {
      int y = iyr[r] + ky, x = ixr[r] + kx;
      bool ok = ((unsigned)y < HW) && ((unsigned)x < HW);
      qf[r] = ok ? fch[bO[r] + y * HW + x] : 0.f;
    }

#pragma unroll
    for (int c = 0; c < 3; c++) {
      /* B fragment from LDS: row g, K {0..7,16..23} (+8 for hi) (ISA 16-bit B layout) */
      const _Float16* bb = &bstage[cur][c][ln][hi * 8];
      v8f acc = {};
#pragma unroll
      for (int ks = 0; ks < 8; ks++) {
        v8h lo = *(const v8h*)(bb + ks * 32);
        v8h hh = *(const v8h*)(bb + ks * 32 + 16);
        v16h Bm = __builtin_shufflevector(lo, hh, 0, 1, 2, 3, 4, 5, 6, 7,
                                          8, 9, 10, 11, 12, 13, 14, 15);
        acc = __builtin_amdgcn_wmma_f32_16x16x32_f16(
            false, A[ks], false, Bm, (short)0, acc, false, false);
      }
      float bias = b2[gg * 3 + c];
#pragma unroll
      for (int r = 0; r < 8; r++)
        part[c][r] += (acc[r] + bias) * qf[r];
    }
    __syncthreads();                           /* reads done before re-issue */
  }

  /* reduce over N (16 lanes within each half), write pred[q][c] */
#pragma unroll
  for (int c = 0; c < 3; c++)
#pragma unroll
    for (int r = 0; r < 8; r++) {
      float v = part[c][r];
      v += __shfl_xor(v, 1, 32);
      v += __shfl_xor(v, 2, 32);
      v += __shfl_xor(v, 4, 32);
      v += __shfl_xor(v, 8, 32);
      if (ln == 0) out[(size_t)(q0 + r + hi * 8) * 3 + c] = v;
    }
}

extern "C" void kernel_launch(void* const* d_in, const int* in_sizes, int n_in,
                              void* d_out, int out_size, void* d_ws, size_t ws_size,
                              hipStream_t stream) {
  (void)in_sizes; (void)n_in; (void)out_size; (void)ws_size;
  const float* inp    = (const float*)d_in[0];
  const float* coord  = (const float*)d_in[1];
  const float* cell   = (const float*)d_in[2];
  const float* conv_w = (const float*)d_in[3];
  const float* conv_b = (const float*)d_in[4];
  const float* fc1_w  = (const float*)d_in[5];
  const float* fc1_b  = (const float*)d_in[6];
  const float* fc2_w  = (const float*)d_in[7];
  const float* fc2_b  = (const float*)d_in[8];

  char* ws = (char*)d_ws;
  float*    feat = (float*)(ws + WS_FEAT);
  _Float16* hdn  = (_Float16*)(ws + WS_HDN);
  _Float16* w2h  = (_Float16*)(ws + WS_W2H);
  int*      qidx = (int*)(ws + WS_QIDX);

  conv3x3_kernel<<<(4 * HW * HW) / 256, 256, 0, stream>>>(inp, conv_w, conv_b, feat);
  prep_w2_kernel<<<(3 * NG * HID + 255) / 256, 256, 0, stream>>>(fc2_w, w2h);
  mlp1_kernel<<<NQ, 256, 0, stream>>>(coord, cell, fc1_w, fc1_b, hdn, qidx);
  meta_sr_fused_kernel<<<NQ / 128, 256, 0, stream>>>(hdn, w2h, fc2_b, feat, qidx,
                                                     (float*)d_out);
}
